// spdRnnNet_16458314678605
// MI455X (gfx1250) — matrix operand order repaired and checked by reading, not compiled, not run on hardware
//
#include <hip/hip_runtime.h>
#include <math.h>

typedef __attribute__((ext_vector_type(16))) _Float16 v16h;
typedef __attribute__((ext_vector_type(8)))  float    v8f;

#define NTHREADS 256
#define NWAVES 8

constexpr int Ss = 800;
constexpr int NLAT = 20, LT = 190, HLs = 63, OD = 83;

// ---------------- workspace layout (bytes) ----------------
constexpr size_t WS_M   = 0;                                   // 22*20 f32
constexpr size_t WS_ACC = 2048;                                // 2*64*96 f32 = 49152 B
constexpr size_t WS_WD  = 51200;                               // [S][64][32] f32
constexpr size_t WS_WL  = WS_WD + (size_t)Ss * 64 * 32 * 4;    // [S][64][192] f32

// ---------------- scan LDS layout ----------------
// f32 region (offsets in floats).  O_LHD holds log(hD) (the natural carry).
constexpr int O_B0 = 0, O_B1 = 96, O_B2 = 144, O_B3 = 240, O_BD = 336, O_BL = 400,
              O_Y0 = 592, O_Y1 = 2128, O_LHD = 3664, O_HL = 4176,
              O_GX = 5200, O_GH = 6224, O_GXL = 7248, O_GHL = 10320, O_AC = 13392,
              F32_COUNT = 14928;                                // 59712 B
// f16 region (offsets in halves), starts at byte F32_COUNT*4
// A-operand buffers (row-major [16][Kpad] f16, fragment load = 2x ds_load_b128)
constexpr int A_Y0 = 0, A_T = 1536, A_U = 3072, A_XD = 4096, A_XL = 4608,
              A_HD = 7680, A_HL = 8192,
// weights, col-major [Npad][Kpad] f16
              H_W0 = 9216, H_W1 = 18432, H_W2 = 23040, H_W3 = 29184,
              H_DX = 38400, H_DH = 40448, H_LX = 42496, H_LH = 79360,
              H_COUNT = 91648;                                  // 183296 B
constexpr size_t SCAN_LDS = (size_t)F32_COUNT * 4 + (size_t)H_COUNT * 2;      // 243008 B
constexpr size_t CHOL_LDS = (size_t)(484 * 64 + 210 * 64 + 440) * 4;          // 179424 B

// ---------------- fast transcendentals (raw v_rcp_f32, no IEEE divide) ----------------
__device__ __forceinline__ float frcp_(float x) { return __builtin_amdgcn_rcpf(x); }
__device__ __forceinline__ float fexp_(float x) { return __expf(x); }
__device__ __forceinline__ float flog_(float x) { return __logf(x); }
__device__ __forceinline__ float ftanh_(float x) {
    float e = __expf(2.f * x);
    return 1.f - 2.f * frcp_(e + 1.f);
}
__device__ __forceinline__ float sigmoidf_(float x) { return frcp_(1.f + __expf(-x)); }
__device__ __forceinline__ float softplusf_(float x) {
    return fmaxf(x, 0.f) + __logf(1.f + __expf(-fabsf(x)));
}

// A fragment: 16x32 f16 (ISA 7.12.2), A stored row-major f16 [16][Kpad].
// lane m=l&15; koff=(l>>4)*8; halves 0..7 -> K=koff..+7, halves 8..15 -> K=16+koff..+7
__device__ __forceinline__ v16h load_a16(const _Float16* A, int Kpad, int kc, int lane) {
    const _Float16* p = A + (lane & 15) * Kpad + kc * 32 + (lane >> 4) * 8;
    v16h a;
#pragma unroll
    for (int j = 0; j < 8; ++j) a[j] = p[j];
#pragma unroll
    for (int j = 0; j < 8; ++j) a[8 + j] = p[16 + j];
    return a;
}

// B fragment: 32x16 f16, W stored col-major [Npad][Kpad] in LDS.
// lane n=l&15; kb=kc*32+(l>>4)*8; halves 0..7 -> K=kb..+7, halves 8..15 -> K=16+kb..+7
__device__ __forceinline__ v16h load_b16(const _Float16* W, int Kpad, int nt, int kc, int lane) {
    const _Float16* p = W + (nt * 16 + (lane & 15)) * Kpad + kc * 32 + (lane >> 4) * 8;
    v16h b;
#pragma unroll
    for (int j = 0; j < 16; ++j) b[j] = p[j + (j & 8)];
    return b;
}

__device__ __forceinline__ v8f wmma16(v16h a, v16h b, v8f c) {
    return __builtin_amdgcn_wmma_f32_16x16x32_f16(false, a, false, b, (short)0, c, false, false);
}

// MODE 0: OUT16 = f16(tanh(acc+bias))  [bias folded into C-init]
// MODE 1: OUT32 = acc + bias + resid   [bias folded into C-init]
// MODE 2: OUT32 = acc                  [raw]
template <int MODE>
__device__ __forceinline__ void gemm16(const _Float16* A, const _Float16* W,
                                       _Float16* OUT16, float* OUT32,
                                       const float* bias, const float* resid,
                                       int Kpad, int Npad, int ostride, int wid, int lane) {
    for (int nt = wid; nt * 16 < Npad; nt += NWAVES) {
        int col = nt * 16 + (lane & 15);
        int rb  = (lane >> 4) * 8;
        v8f acc;
        if (MODE == 2) {
            acc = (v8f){0.f, 0.f, 0.f, 0.f, 0.f, 0.f, 0.f, 0.f};
        } else {
            float bv = bias[col];
#pragma unroll
            for (int r = 0; r < 8; ++r) acc[r] = bv;
        }
        int nkc = Kpad >> 5;
        for (int kc = 0; kc < nkc; ++kc)
            acc = wmma16(load_a16(A, Kpad, kc, lane), load_b16(W, Kpad, nt, kc, lane), acc);
#pragma unroll
        for (int r = 0; r < 8; ++r) {
            if (MODE == 0)      OUT16[(rb + r) * ostride + col] = (_Float16)ftanh_(acc[r]);
            else if (MODE == 1) OUT32[(rb + r) * ostride + col] = acc[r] + resid[(rb + r) * ostride + col];
            else                OUT32[(rb + r) * ostride + col] = acc[r];
        }
    }
}

__device__ __forceinline__ void pack_w(_Float16* dst, const float* src, int Kv, int Nv,
                                       int Kp, int Np, bool do_abs, int tid) {
    for (int i = tid; i < Kp * Np; i += NTHREADS) {
        int n = i / Kp, k = i % Kp;
        float v = (k < Kv && n < Nv) ? src[k * Nv + n] : 0.f;
        if (do_abs) v = fabsf(v);
        dst[i] = (_Float16)v;
    }
}
__device__ __forceinline__ void pack_b(float* dst, const float* src, int Nv, int Np,
                                       bool do_abs, int tid) {
    for (int i = tid; i < Np; i += NTHREADS) {
        float v = (i < Nv) ? src[i] : 0.f;
        dst[i] = do_abs ? fabsf(v) : v;
    }
}

// ---------------- kernel 1: M = W_bimap1 @ W_bimap2 (22x22 @ 22x20) ----------------
__global__ void k_matM(const float* __restrict__ W1g, const float* __restrict__ W2g,
                       float* __restrict__ Mg) {
    for (int o = threadIdx.x; o < 22 * 20; o += NTHREADS) {
        int i = o / 20, j = o % 20;
        float s = 0.f;
#pragma unroll
        for (int k = 0; k < 22; ++k) s += W1g[i * 22 + k] * W2g[k * 20 + j];
        Mg[o] = s;
    }
}

// ---------------- kernel 2: bilinear map + Cholesky, 64 items per block ----------------
__global__ void __launch_bounds__(64) k_chol(const float* __restrict__ xg,
                                             const float* __restrict__ Mg,
                                             float* __restrict__ wD, float* __restrict__ wL) {
    extern __shared__ float sm[];
    float* sx = sm;                        // [484][64]
    float* sy = sm + 484 * 64;             // [210][64] lower-tri, idx(i,j)=i*(i+1)/2+j
    float* sM = sm + (484 + 210) * 64;     // [440]
    const int tid = threadIdx.x;

    size_t base = (size_t)blockIdx.x * 64 * 484;
    for (int i = tid; i < 64 * 484; i += 64) {
        int it = i / 484, e = i % 484;
        sx[e * 64 + it] = xg[base + i];
    }
    for (int i = tid; i < 440; i += 64) sM[i] = Mg[i];
    __syncthreads();

    // y = M^T x M, lower triangle (x is symmetric)
    for (int j = 0; j < 20; ++j) {
        float Mj[22];
#pragma unroll
        for (int l = 0; l < 22; ++l) Mj[l] = sM[l * 20 + j];
        float v[22];
#pragma unroll
        for (int k = 0; k < 22; ++k) {
            float a = 0.f;
#pragma unroll
            for (int l = 0; l < 22; ++l) a += sx[(k * 22 + l) * 64 + tid] * Mj[l];
            v[k] = a;
        }
        for (int i = j; i < 20; ++i) {
            float a = 0.f;
#pragma unroll
            for (int k = 0; k < 22; ++k) a += sM[k * 20 + i] * v[k];
            sy[(i * (i + 1) / 2 + j) * 64 + tid] = a;
        }
    }
    // in-place Cholesky (per-thread column in LDS; keep IEEE divide for accuracy)
    for (int j = 0; j < 20; ++j) {
        float s = sy[(j * (j + 1) / 2 + j) * 64 + tid];
        for (int k = 0; k < j; ++k) {
            float ljk = sy[(j * (j + 1) / 2 + k) * 64 + tid];
            s -= ljk * ljk;
        }
        float d = sqrtf(s);
        sy[(j * (j + 1) / 2 + j) * 64 + tid] = d;
        float inv = 1.f / d;
        for (int i = j + 1; i < 20; ++i) {
            float s2 = sy[(i * (i + 1) / 2 + j) * 64 + tid];
            for (int k = 0; k < j; ++k)
                s2 -= sy[(i * (i + 1) / 2 + k) * 64 + tid] * sy[(j * (j + 1) / 2 + k) * 64 + tid];
            sy[(i * (i + 1) / 2 + j) * 64 + tid] = s2 * inv;
        }
    }
    // write padded [S][64][32] diag and [S][64][192] lower-tri
    int item = blockIdx.x * 64 + tid;
    int b = item / Ss, s = item % Ss;
    size_t dbase = ((size_t)s * 64 + b) * 32;
    for (int d = 0; d < 20; ++d) wD[dbase + d] = sy[(d * (d + 1) / 2 + d) * 64 + tid];
    for (int d = 20; d < 32; ++d) wD[dbase + d] = 0.f;
    size_t lbase = ((size_t)s * 64 + b) * 192;
    int e = 0;
    for (int i = 1; i < 20; ++i)
        for (int j2 = 0; j2 < i; ++j2)
            wL[lbase + (e++)] = sy[(i * (i + 1) / 2 + j2) * 64 + tid];
    wL[lbase + 190] = 0.f;
    wL[lbase + 191] = 0.f;
}

// ---------------- kernel 3: the 800-step scan ----------------
struct ScanParams {
    const float *W0[2], *b0[2], *W1[2], *b1[2], *W2[2], *b2[2], *W3[2], *b3[2];
    const float *DWx[2], *DWh[2], *Db[2];
    const float *LWx[2], *LWh[2], *Lb[2];
    const float *wD, *wL;
    float* acc;   // [2][64][96]
};

__global__ void __launch_bounds__(NTHREADS) k_scan(ScanParams P) {
    extern __shared__ char smem[];
    float* F = (float*)smem;
    _Float16* H = (_Float16*)(smem + (size_t)F32_COUNT * 4);
    const int tid  = threadIdx.x;
    const int wid  = tid >> 5, lane = tid & 31;
    const int st   = blockIdx.x >> 2;          // 0=fwd, 1=rev
    const int rb16 = (blockIdx.x & 3) * 16;    // batch-row base

    // pack weights to f16 col-major [Npad][Kpad] (zero padded), biases (abs where needed)
    pack_w(H + H_W0, P.W0[st], OD, OD, 96, 96, false, tid);
    pack_w(H + H_W1, P.W1[st], OD, 41, 96, 48, false, tid);
    pack_w(H + H_W2, P.W2[st], 41, OD, 64, 96, false, tid);
    pack_w(H + H_W3, P.W3[st], OD, OD, 96, 96, false, tid);
    pack_w(H + H_DX, P.DWx[st], NLAT, 60, 32, 64, true, tid);
    pack_w(H + H_DH, P.DWh[st], NLAT, 60, 32, 64, true, tid);
    pack_w(H + H_LX, P.LWx[st], LT, 189, 192, 192, false, tid);
    pack_w(H + H_LH, P.LWh[st], HLs, 189, 64, 192, false, tid);
    pack_b(F + O_B0, P.b0[st], OD, 96, false, tid);
    pack_b(F + O_B1, P.b1[st], 41, 48, false, tid);
    pack_b(F + O_B2, P.b2[st], OD, 96, false, tid);
    pack_b(F + O_B3, P.b3[st], OD, 96, false, tid);
    pack_b(F + O_BD, P.Db[st], 60, 64, true, tid);
    pack_b(F + O_BL, P.Lb[st], 189, 192, false, tid);
    // carries: log(hD)=log(1)=0, hL=0
    for (int i = tid; i < 512;  i += NTHREADS) F[O_LHD + i] = 0.f;
    for (int i = tid; i < 1024; i += NTHREADS) F[O_HL + i] = 0.f;
    for (int i = tid; i < 1536; i += NTHREADS) { F[O_AC + i] = 0.f; F[O_Y0 + i] = 0.f; }
    // zero f16 A-buffers whose pads must stay 0 across the scan
    for (int i = tid; i < 1536; i += NTHREADS) H[A_Y0 + i] = (_Float16)0.f;
    for (int i = tid; i < 512;  i += NTHREADS) H[A_HD + i] = (_Float16)0.f;
    for (int i = tid; i < 1024; i += NTHREADS) H[A_HL + i] = (_Float16)0.f;
    __syncthreads();

    const float* gD = P.wD;
    const float* gL = P.wL;

    for (int t = 0; t < Ss; ++t) {
        int sidx = st ? (Ss - 1 - t) : t;
        // issue this step's input loads early (consumed after the ODE phase)
        size_t dbase = ((size_t)sidx * 64 + rb16) * 32;
        size_t lbase = ((size_t)sidx * 64 + rb16) * 192;
        float xd0 = gD[dbase + tid];
        float xd1 = gD[dbase + 256 + tid];
        float xl[12];
#pragma unroll
        for (int i = 0; i < 12; ++i) xl[i] = gL[lbase + i * 256 + tid];
        if (t + 1 < Ss) {
            int s2 = st ? (Ss - 2 - t) : (t + 1);
            __builtin_prefetch(&gL[((size_t)s2 * 64 + rb16) * 192 + (size_t)tid * 12], 0, 1);
        }

        // (1) Y0 = [log hD | hL] -- log hD is the carry itself
        for (int i = tid; i < 16 * 83; i += NTHREADS) {
            int row = i / 83, c = i % 83;
            float v = (c < 20) ? F[O_LHD + row * 32 + c] : F[O_HL + row * 64 + (c - 20)];
            F[O_Y0 + row * 96 + c] = v;
            H[A_Y0 + row * 96 + c] = (_Float16)v;
        }
        __syncthreads();
        // (2) ODE MLP (WMMA), f16 intermediates
        gemm16<0>(H + A_Y0, H + H_W0, H + A_T, nullptr, F + O_B0, nullptr, 96, 96, 96, wid, lane);
        __syncthreads();
        gemm16<0>(H + A_T,  H + H_W1, H + A_U, nullptr, F + O_B1, nullptr, 96, 48, 64, wid, lane);
        __syncthreads();
        gemm16<0>(H + A_U,  H + H_W2, H + A_T, nullptr, F + O_B2, nullptr, 64, 96, 96, wid, lane);
        __syncthreads();
        gemm16<1>(H + A_T,  H + H_W3, nullptr, F + O_Y1, F + O_B3, F + O_Y0, 96, 96, 96, wid, lane);
        __syncthreads();
        // (3) extract carries: log hD = tanh(y1[:20]); hD (f16, for GH matmul) = exp of it
        for (int i = tid; i < 16 * 83; i += NTHREADS) {
            int row = i / 83, c = i % 83;
            float v = F[O_Y1 + row * 96 + c];
            if (c < 20) {
                float tt = ftanh_(v);
                F[O_LHD + row * 32 + c] = tt;
                H[A_HD + row * 32 + c] = (_Float16)fexp_(tt);
            } else {
                F[O_HL + row * 64 + (c - 20)] = v;
                H[A_HL + row * 64 + (c - 20)] = (_Float16)v;
            }
        }
        H[A_XD + tid] = (_Float16)xd0;
        H[A_XD + 256 + tid] = (_Float16)xd1;
#pragma unroll
        for (int i = 0; i < 12; ++i) H[A_XL + i * 256 + tid] = (_Float16)xl[i];
        __syncthreads();
        // (4) RGRU matmuls (WMMA): GX=xD@|Wx|, GH=hD@|Wh|, GXL=xL@Wx, GHL=hL@Wh
        for (int job = wid; job < 32; job += NWAVES) {
            const _Float16* A; const _Float16* W; float* OUT; int Kp, nt, ostr;
            if (job < 4)       { A = H + A_XD; W = H + H_DX; OUT = F + O_GX;  Kp = 32;  nt = job;      ostr = 64;  }
            else if (job < 8)  { A = H + A_HD; W = H + H_DH; OUT = F + O_GH;  Kp = 32;  nt = job - 4;  ostr = 64;  }
            else if (job < 20) { A = H + A_XL; W = H + H_LX; OUT = F + O_GXL; Kp = 192; nt = job - 8;  ostr = 192; }
            else               { A = H + A_HL; W = H + H_LH; OUT = F + O_GHL; Kp = 64;  nt = job - 20; ostr = 192; }
            v8f acc = {};
            int nkc = Kp >> 5;
            for (int kc = 0; kc < nkc; ++kc)
                acc = wmma16(load_a16(A, Kp, kc, lane), load_b16(W, Kp, nt, kc, lane), acc);
            int col = nt * 16 + (lane & 15), rbs = (lane >> 4) * 8;
#pragma unroll
            for (int r = 0; r < 8; ++r) OUT[(rbs + r) * ostr + col] = acc[r];
        }
        __syncthreads();
        // (5) RGRU combines; new carries + out_t accumulation fused (out = [log hD | hL])
        for (int i = tid; i < 16 * 20; i += NTHREADS) {
            int row = i / 20, c = i % 20;
            float iR = F[O_GX + row * 64 + c], iI = F[O_GX + row * 64 + 20 + c], iN = F[O_GX + row * 64 + 40 + c];
            float hR = F[O_GH + row * 64 + c], hI = F[O_GH + row * 64 + 20 + c], hN = F[O_GH + row * 64 + 40 + c];
            float r = sigmoidf_(F[O_BD + c] * iR * hR);
            float z = sigmoidf_(F[O_BD + 20 + c] * iI * hI);
            float n = softplusf_(F[O_BD + 40 + c] * iN * (r * hN));
            float lho = F[O_LHD + row * 32 + c];                 // log of ODE-updated hD
            float q = z * flog_(n) + (1.f - z) * lho;            // log of new hD
            F[O_LHD + row * 32 + c] = q;
            F[O_AC + row * 96 + c] += q;
        }
        for (int i = tid; i < 16 * 63; i += NTHREADS) {
            int row = i / 63, c = i % 63;
            float iR = F[O_GXL + row * 192 + c], iI = F[O_GXL + row * 192 + 63 + c], iN = F[O_GXL + row * 192 + 126 + c];
            float hR = F[O_GHL + row * 192 + c], hI = F[O_GHL + row * 192 + 63 + c], hN = F[O_GHL + row * 192 + 126 + c];
            float r = sigmoidf_(iR + hR + F[O_BL + c]);
            float z = sigmoidf_(iI + hI + F[O_BL + 63 + c]);
            float n = ftanh_(iN + r * hN + F[O_BL + 126 + c]);
            float ho = F[O_HL + row * 64 + c];
            float hn = (1.f - z) * ho + z * n;
            F[O_HL + row * 64 + c] = hn;
            F[O_AC + row * 96 + 20 + c] += hn;
        }
        __syncthreads();
    }

    for (int i = tid; i < 16 * 83; i += NTHREADS) {
        int row = i / 83, c = i % 83;
        P.acc[((size_t)(st * 64 + rb16 + row)) * 96 + c] = F[O_AC + row * 96 + c] * (1.f / (float)Ss);
    }
}

// ---------------- kernel 4: classifier ----------------
__global__ void k_cls(const float* __restrict__ acc, const float* __restrict__ Wg,
                      const float* __restrict__ bg, float* __restrict__ out) {
    for (int o = threadIdx.x; o < 64 * 40; o += NTHREADS) {
        int b = o / 40, c = o % 40;
        float s = bg[c];
        for (int k = 0; k < 83; ++k)
            s += (acc[(size_t)b * 96 + k] + acc[(size_t)(64 + b) * 96 + k]) * Wg[k * 40 + c];
        out[o] = s;
    }
}

// ---------------- host ----------------
extern "C" void kernel_launch(void* const* d_in, const int* in_sizes, int n_in,
                              void* d_out, int out_size, void* d_ws, size_t ws_size,
                              hipStream_t stream) {
    (void)in_sizes; (void)n_in; (void)out_size; (void)ws_size;
    const float* x   = (const float*)d_in[0];
    const float* Wb1 = (const float*)d_in[1];
    const float* Wb2 = (const float*)d_in[2];

    ScanParams P;
    P.W0[0] = (const float*)d_in[3];  P.b0[0] = (const float*)d_in[4];
    P.W1[0] = (const float*)d_in[5];  P.b1[0] = (const float*)d_in[6];
    P.W2[0] = (const float*)d_in[7];  P.b2[0] = (const float*)d_in[8];
    P.W3[0] = (const float*)d_in[9];  P.b3[0] = (const float*)d_in[10];
    P.W0[1] = (const float*)d_in[11]; P.b0[1] = (const float*)d_in[12];
    P.W1[1] = (const float*)d_in[13]; P.b1[1] = (const float*)d_in[14];
    P.W2[1] = (const float*)d_in[15]; P.b2[1] = (const float*)d_in[16];
    P.W3[1] = (const float*)d_in[17]; P.b3[1] = (const float*)d_in[18];
    P.DWx[0] = (const float*)d_in[19]; P.DWh[0] = (const float*)d_in[20]; P.Db[0] = (const float*)d_in[21];
    P.LWx[0] = (const float*)d_in[22]; P.LWh[0] = (const float*)d_in[23]; P.Lb[0] = (const float*)d_in[24];
    P.DWx[1] = (const float*)d_in[25]; P.DWh[1] = (const float*)d_in[26]; P.Db[1] = (const float*)d_in[27];
    P.LWx[1] = (const float*)d_in[28]; P.LWh[1] = (const float*)d_in[29]; P.Lb[1] = (const float*)d_in[30];
    const float* clsW = (const float*)d_in[31];
    const float* clsb = (const float*)d_in[32];

    char* ws = (char*)d_ws;
    float* Mg  = (float*)(ws + WS_M);
    float* acc = (float*)(ws + WS_ACC);
    float* wD  = (float*)(ws + WS_WD);
    float* wL  = (float*)(ws + WS_WL);
    P.wD = wD; P.wL = wL; P.acc = acc;

    k_matM<<<1, NTHREADS, 0, stream>>>(Wb1, Wb2, Mg);
    k_chol<<<Ss, 64, CHOL_LDS, stream>>>(x, Mg, wD, wL);
    k_scan<<<8, NTHREADS, SCAN_LDS, stream>>>(P);
    k_cls<<<1, NTHREADS, 0, stream>>>(acc, clsW, clsb, (float*)d_out);
}
